// EfficientCrossAttention_11141145166514
// MI455X (gfx1250) — matrix operand order repaired and checked by reading, not compiled, not run on hardware
//
#include <hip/hip_runtime.h>
#include <hip/hip_bf16.h>

#define DIMX      1024
#define HEADS     16
#define DIM_HEAD  64
#define NQ        2048
#define NKV       2048
#define BATCH     2

typedef __attribute__((ext_vector_type(16))) _Float16 v16h;
typedef __attribute__((ext_vector_type(8)))  _Float16 v8h;
typedef __attribute__((ext_vector_type(8)))  float    v8f;
typedef __attribute__((ext_vector_type(4)))  int      v4i;

// ---------------- CDNA5 async global->LDS copy (ASYNCcnt path) -------------
#if defined(__has_builtin)
#  if __has_builtin(__builtin_amdgcn_global_load_async_to_lds_b128) && \
      __has_builtin(__builtin_amdgcn_s_wait_asynccnt)
#    define USE_ASYNC_LDS 1
#  endif
#endif
#ifndef USE_ASYNC_LDS
#  define USE_ASYNC_LDS 0
#endif

#if USE_ASYNC_LDS
typedef __attribute__((address_space(1))) v4i v4i_g;   // global
typedef __attribute__((address_space(3))) v4i v4i_l;   // LDS
#endif

__device__ __forceinline__ void cp16_g2l(const _Float16* g, _Float16* l) {
#if USE_ASYNC_LDS
  __builtin_amdgcn_global_load_async_to_lds_b128((v4i_g*)g, (v4i_l*)l, 0, 0);
#else
  *(v8h*)l = *(const v8h*)g;
#endif
}

__device__ __forceinline__ void async_wait_all() {
#if USE_ASYNC_LDS
  __builtin_amdgcn_s_wait_asynccnt(0);
#endif
}

__device__ __forceinline__ v8f wmma_f32_f16(v16h a, v16h b, v8f c) {
  return __builtin_amdgcn_wmma_f32_16x16x32_f16(false, a, false, b, (short)0, c,
                                                false, false);
}

// A-operand fragment (16x32 f16, MxK). Lane holds row lane%16.
// lanes 0-15: K {0..7} in r[0..7], K {16..23} in r[8..15]
// lanes16-31: K {8..15},            K {24..31}
// caller passes p = base + row*ldk + (lane>>4)*8
__device__ __forceinline__ v16h ld_fragA(const _Float16* p) {
  v8h lo = *(const v8h*)p;
  v8h hi = *(const v8h*)(p + 16);
  v16h r;
#pragma unroll
  for (int i = 0; i < 8; ++i) { r[i] = lo[i]; r[i + 8] = hi[i]; }
  return r;
}

// B-operand fragment (32x16 f16, KxN). Lane holds column lane%16.
// lanes 0-15: K {0..7},{16..23}; lanes 16-31: K {8..15},{24..31}.
// caller passes p = baseT + col*ldk + (lane>>4)*8   (B stored transposed [N][K])
__device__ __forceinline__ v16h ld_fragB(const _Float16* p) {
  v8h lo = *(const v8h*)p;
  v8h hi = *(const v8h*)(p + 16);
  v16h r;
#pragma unroll
  for (int i = 0; i < 8; ++i) { r[i] = lo[i]; r[i + 8] = hi[i]; }
  return r;
}

// ---------------- prep kernels ----------------

__global__ void cvt_f32_f16(const float* __restrict__ in,
                            _Float16* __restrict__ out, size_t n) {
  size_t i = (size_t)blockIdx.x * blockDim.x + threadIdx.x;
  if (i < n) out[i] = (_Float16)in[i];
}

// out[n*K + k] = (f16) in[k*N + n]
__global__ void transpose_f16(const float* __restrict__ in,
                              _Float16* __restrict__ out, int K, int N) {
  size_t i = (size_t)blockIdx.x * blockDim.x + threadIdx.x;
  if (i < (size_t)K * N) {
    int k = (int)(i / N);
    int n = (int)(i - (size_t)k * N);
    out[(size_t)n * K + k] = (_Float16)in[i];
  }
}

__global__ __launch_bounds__(256) void ln_f16(const float* __restrict__ x,
                                              const float* __restrict__ g,
                                              const float* __restrict__ bb,
                                              _Float16* __restrict__ out, int C) {
  __shared__ float red[256];
  int row = blockIdx.x;
  const float* xr = x + (size_t)row * C;
  float s = 0.f;
  for (int i = threadIdx.x; i < C; i += 256) s += xr[i];
  red[threadIdx.x] = s; __syncthreads();
  for (int st = 128; st > 0; st >>= 1) {
    if ((int)threadIdx.x < st) red[threadIdx.x] += red[threadIdx.x + st];
    __syncthreads();
  }
  float mu = red[0] / (float)C;
  __syncthreads();
  float sq = 0.f;
  for (int i = threadIdx.x; i < C; i += 256) { float d = xr[i] - mu; sq += d * d; }
  red[threadIdx.x] = sq; __syncthreads();
  for (int st = 128; st > 0; st >>= 1) {
    if ((int)threadIdx.x < st) red[threadIdx.x] += red[threadIdx.x + st];
    __syncthreads();
  }
  float rs = rsqrtf(red[0] / (float)C + 1e-5f);
  _Float16* orow = out + (size_t)row * C;
  for (int i = threadIdx.x; i < C; i += 256)
    orow[i] = (_Float16)((xr[i] - mu) * rs * g[i] + bb[i]);
}

// ---------------- WMMA GEMM: C = A[M,K] * Bt[N,K]^T ----------------
// one wave per 32(M) x 64(N) tile, fp32 accumulate.
// MODE 0: store f16, split heads -> q[b,h,n,d], fold `scale`
// MODE 1: n<DIMX -> k[b,h,n,d] ; n>=DIMX -> vT[b,h,d,n]
// MODE 2: store f32 row-major [M,N]
template <int MODE>
__global__ __launch_bounds__(32) void gemm_wmma(
    const _Float16* __restrict__ A, const _Float16* __restrict__ Bt,
    void* __restrict__ C0, void* __restrict__ C1,
    int M, int N, int K, int rowsPerBatch, float scale) {
  const int lane = threadIdx.x & 31;
  const int lrow = lane & 15;
  const int lhi  = lane >> 4;
  const int m0 = blockIdx.x * 32;
  const int n0 = blockIdx.y * 64;

  v8f acc[2][4] = {};

  const _Float16* a0p = A + (size_t)(m0 + lrow) * K + lhi * 8;
  const _Float16* a1p = A + (size_t)(m0 + 16 + lrow) * K + lhi * 8;
  const _Float16* bp[4];
#pragma unroll
  for (int t = 0; t < 4; ++t)
    bp[t] = Bt + (size_t)(n0 + t * 16 + lrow) * K + lhi * 8;

  for (int kk = 0; kk < K; kk += 32) {
    v16h a0 = ld_fragA(a0p + kk);
    v16h a1 = ld_fragA(a1p + kk);
#pragma unroll
    for (int t = 0; t < 4; ++t) {
      v16h b = ld_fragB(bp[t] + kk);
      acc[0][t] = wmma_f32_f16(a0, b, acc[0][t]);
      acc[1][t] = wmma_f32_f16(a1, b, acc[1][t]);
    }
  }

#pragma unroll
  for (int mi = 0; mi < 2; ++mi) {
#pragma unroll
    for (int t = 0; t < 4; ++t) {
#pragma unroll
      for (int j = 0; j < 8; ++j) {
        int r = m0 + mi * 16 + j + lhi * 8;
        int n = n0 + t * 16 + lrow;
        float v = acc[mi][t][j];
        if (MODE == 2) {
          ((float*)C0)[(size_t)r * N + n] = v;
        } else {
          int b  = r / rowsPerBatch;
          int rr = r - b * rowsPerBatch;
          if (MODE == 0) {
            int h = n >> 6, d = n & 63;
            ((_Float16*)C0)[(((size_t)b * HEADS + h) * rowsPerBatch + rr) * DIM_HEAD + d] =
                (_Float16)(v * scale);
          } else {
            if (n < DIMX) {
              int h = n >> 6, d = n & 63;
              ((_Float16*)C0)[(((size_t)b * HEADS + h) * rowsPerBatch + rr) * DIM_HEAD + d] =
                  (_Float16)v;
            } else {
              int n2 = n - DIMX;
              int h = n2 >> 6, d = n2 & 63;
              ((_Float16*)C1)[(((size_t)b * HEADS + h) * DIM_HEAD + d) * rowsPerBatch + rr] =
                  (_Float16)v;
            }
          }
        }
      }
    }
  }
}

// ---------------- flash attention ----------------
// 4 waves / block, 64 q rows per block (16 per wave). K/V staged into LDS
// in 64-column chunks (double-buffered) with async global->LDS copies shared
// by all 4 waves.
// q: [b,h,nq,64] (softmax scale pre-folded), k: [b,h,nk,64], vT: [b,h,64,nk]
// omid: [b*nq, 1024] f16 (heads merged)
__global__ __launch_bounds__(128) void attn_wmma(
    const _Float16* __restrict__ q, const _Float16* __restrict__ k,
    const _Float16* __restrict__ vT, _Float16* __restrict__ omid,
    int Nq, int Nk, int H) {
  __shared__ __align__(16) _Float16 kbuf[2][64 * 64];   // [kv row][d]
  __shared__ __align__(16) _Float16 vbuf[2][64 * 64];   // [d][kv row]
  __shared__ __align__(16) _Float16 pbuf[4][16 * 64];   // per-wave P staging

  const int tid  = threadIdx.x;
  const int wave = tid >> 5;
  const int lane = tid & 31;
  const int lrow = lane & 15;
  const int lhi  = lane >> 4;

  const int qblocks = Nq >> 6;               // 64 q rows per block
  const int qt = blockIdx.x % qblocks;
  const int bh = blockIdx.x / qblocks;       // b*H + h

  const _Float16* qp = q  + ((size_t)bh * Nq + qt * 64 + wave * 16) * DIM_HEAD;
  const _Float16* kp = k  + (size_t)bh * Nk * DIM_HEAD;
  const _Float16* vp = vT + (size_t)bh * DIM_HEAD * Nk;

  // Q tile 16x64 as two A fragments (K = 0..31, 32..63)
  v16h aq0 = ld_fragA(qp + (size_t)lrow * DIM_HEAD + 0  + lhi * 8);
  v16h aq1 = ld_fragA(qp + (size_t)lrow * DIM_HEAD + 32 + lhi * 8);

  float mrow[8], lsum[8];
#pragma unroll
  for (int j = 0; j < 8; ++j) { mrow[j] = -1e30f; lsum[j] = 0.f; }
  v8f acc[4] = {};

  // stage a 64-column K/V chunk: K chunk is 8KB contiguous; V chunk is 64
  // rows of 128B. 128 threads x (4+4) b128 copies.
  auto stage = [&](int jj, int buf) {
    const _Float16* gk = kp + (size_t)jj * DIM_HEAD;     // 64*64 contiguous
    _Float16* lk = kbuf[buf];
    _Float16* lv = vbuf[buf];
#pragma unroll
    for (int i = 0; i < 4; ++i) {
      int c = tid + 128 * i;                 // 512 chunks of 8 halves
      cp16_g2l(gk + (size_t)c * 8, lk + c * 8);
      int d = c >> 3, koff = (c & 7) * 8;    // c*8 == d*64 + koff
      cp16_g2l(vp + (size_t)d * Nk + jj + koff, lv + c * 8);
    }
  };

  stage(0, 0);
  int buf = 0;
  for (int jj = 0; jj < Nk; jj += 64) {
    async_wait_all();
    __syncthreads();                         // chunk `buf` ready for all waves
    if (jj + 64 < Nk) stage(jj + 64, buf ^ 1);

    const _Float16* kb = kbuf[buf];
    const _Float16* vb = vbuf[buf];

    // S = Q * K^T for 4 column tiles of 16
    v8f s[4];
#pragma unroll
    for (int jt = 0; jt < 4; ++jt) {
      v8f z = {};
      const _Float16* kr = &kb[(jt * 16 + lrow) * 64 + lhi * 8];
      z = wmma_f32_f16(aq0, ld_fragB(kr), z);
      z = wmma_f32_f16(aq1, ld_fragB(kr + 32), z);
      s[jt] = z;
    }

    // online softmax across the 64 new columns; row = j + lhi*8
    _Float16* pb = pbuf[wave];
#pragma unroll
    for (int j = 0; j < 8; ++j) {
      float mx = fmaxf(fmaxf(s[0][j], s[1][j]), fmaxf(s[2][j], s[3][j]));
#pragma unroll
      for (int m = 8; m >= 1; m >>= 1) mx = fmaxf(mx, __shfl_xor(mx, m, 32));
      float mnew  = fmaxf(mrow[j], mx);
      float alpha = __expf(mrow[j] - mnew);
      float p0 = __expf(s[0][j] - mnew);
      float p1 = __expf(s[1][j] - mnew);
      float p2 = __expf(s[2][j] - mnew);
      float p3 = __expf(s[3][j] - mnew);
      float ls = (p0 + p1) + (p2 + p3);
#pragma unroll
      for (int m = 8; m >= 1; m >>= 1) ls += __shfl_xor(ls, m, 32);
      lsum[j] = lsum[j] * alpha + ls;
      mrow[j] = mnew;
#pragma unroll
      for (int t = 0; t < 4; ++t) acc[t][j] *= alpha;
      int r = j + lhi * 8;
      pb[r * 64 + 0  + lrow] = (_Float16)p0;
      pb[r * 64 + 16 + lrow] = (_Float16)p1;
      pb[r * 64 + 32 + lrow] = (_Float16)p2;
      pb[r * 64 + 48 + lrow] = (_Float16)p3;
    }
    __syncthreads();                         // order pbuf write -> fragment read

    // O += P(16x64) * V(64x64): P as two A fragments, V columns from LDS
    v16h ap0 = ld_fragA(&pb[lrow * 64 + 0  + lhi * 8]);
    v16h ap1 = ld_fragA(&pb[lrow * 64 + 32 + lhi * 8]);
#pragma unroll
    for (int t = 0; t < 4; ++t) {
      const _Float16* vr = &vb[(t * 16 + lrow) * 64 + lhi * 8];
      acc[t] = wmma_f32_f16(ap0, ld_fragB(vr), acc[t]);
      acc[t] = wmma_f32_f16(ap1, ld_fragB(vr + 32), acc[t]);
    }
    __syncthreads();                         // all reads of `buf` done
    buf ^= 1;
  }

  const int b = bh / H, h = bh - b * H;
#pragma unroll
  for (int j = 0; j < 8; ++j) {
    float inv = 1.0f / lsum[j];
    int r = qt * 64 + wave * 16 + j + lhi * 8;
    size_t base = ((size_t)b * Nq + r) * (size_t)(H * DIM_HEAD) + (size_t)h * DIM_HEAD;
#pragma unroll
    for (int t = 0; t < 4; ++t)
      omid[base + t * 16 + lrow] = (_Float16)(acc[t][j] * inv);
  }
}

// ---------------- launcher ----------------

extern "C" void kernel_launch(void* const* d_in, const int* in_sizes, int n_in,
                              void* d_out, int out_size, void* d_ws, size_t ws_size,
                              hipStream_t stream) {
  (void)in_sizes; (void)n_in; (void)out_size; (void)ws_size;

  const float* x    = (const float*)d_in[0];
  const float* ctx  = (const float*)d_in[1];
  const float* Wq   = (const float*)d_in[2];
  const float* Wkv  = (const float*)d_in[3];
  const float* Wo   = (const float*)d_in[4];
  const float* ln_g = (const float*)d_in[5];
  const float* ln_b = (const float*)d_in[6];
  float* out = (float*)d_out;

  size_t off = 0;
  auto carve = [&](size_t bytes) -> void* {
    void* p = (char*)d_ws + off;
    off += (bytes + 255) & ~(size_t)255;
    return p;
  };
  const size_t tokens = (size_t)BATCH * NQ;  // 4096
  _Float16* xh   = (_Float16*)carve(tokens * DIMX * 2);
  _Float16* ch   = (_Float16*)carve(tokens * DIMX * 2);
  _Float16* WqT  = (_Float16*)carve((size_t)DIMX * DIMX * 2);
  _Float16* WkvT = (_Float16*)carve((size_t)DIMX * 2 * DIMX * 2);
  _Float16* WoT  = (_Float16*)carve((size_t)DIMX * DIMX * 2);
  _Float16* qh   = (_Float16*)carve(tokens * DIMX * 2);
  _Float16* kh   = (_Float16*)carve(tokens * DIMX * 2);
  _Float16* vTh  = (_Float16*)carve(tokens * DIMX * 2);
  _Float16* omid = (_Float16*)carve(tokens * DIMX * 2);

  cvt_f32_f16<<<(unsigned)((tokens * DIMX) / 256), 256, 0, stream>>>(x, xh, tokens * DIMX);
  ln_f16<<<(unsigned)tokens, 256, 0, stream>>>(ctx, ln_g, ln_b, ch, DIMX);
  transpose_f16<<<(DIMX * DIMX) / 256, 256, 0, stream>>>(Wq, WqT, DIMX, DIMX);
  transpose_f16<<<(DIMX * 2 * DIMX) / 256, 256, 0, stream>>>(Wkv, WkvT, DIMX, 2 * DIMX);
  transpose_f16<<<(DIMX * DIMX) / 256, 256, 0, stream>>>(Wo, WoT, DIMX, DIMX);

  dim3 gq((unsigned)(tokens / 32), DIMX / 64);
  gemm_wmma<0><<<gq, 32, 0, stream>>>(xh, WqT, qh, nullptr,
                                      (int)tokens, DIMX, DIMX, NQ, 0.125f);
  dim3 gkv((unsigned)(tokens / 32), (2 * DIMX) / 64);
  gemm_wmma<1><<<gkv, 32, 0, stream>>>(ch, WkvT, kh, vTh,
                                       (int)tokens, 2 * DIMX, DIMX, NKV, 1.0f);

  attn_wmma<<<BATCH * HEADS * (NQ / 64), 128, 0, stream>>>(qh, kh, vTh, omid,
                                                           NQ, NKV, HEADS);

  dim3 go((unsigned)(tokens / 32), DIMX / 64);
  gemm_wmma<2><<<go, 32, 0, stream>>>(omid, WoT, out, nullptr,
                                      (int)tokens, DIMX, DIMX, NQ, 1.0f);
}
